// SocialModel_13675175871130
// MI455X (gfx1250) — matrix operand order repaired and checked by reading, not compiled, not run on hardware
//
#include <hip/hip_runtime.h>
#include <hip/hip_bf16.h>

// Social-LSTM on MI455X (gfx1250, wave32, WMMA bf16 16x16x32).
//
// Per-step pipeline (stream-ordered, 20 steps):
//   k_einsum : st[p,g,r] = sum_n grid[p,n,g] * h[n,r]   (dominant GEMM, WMMA,
//              2 peds/WG, double-buffered LDS A staging, grid prefetch)
//   k_g1     : te = relu(st @ W_t + b_t)  (WMMA),  ie = relu(pos@W_in+b_in)
//   k_g2     : gates = [ie|te|h] @ [W_ih|W_hh]^T + b    (WMMA)
//   k_lstm   : pointwise LSTM cell, h/c update, bf16 h copies, W_out head
//
// Roofline: grids tensor (134 MB over obs phase) is the only mandatory HBM
// traffic (~5.7us at 23.3 TB/s); pred phase reuses the 16.8 MB t=7 slice from
// the 192 MB L2. Sharing each B fragment across 2 peds halves the h-matrix
// L2 traffic (64->32 MB/step) and barriers per WMMA.

#define NPED 512
#define RNN  128
#define G2C  16
#define EMB  64
#define OUTC 5
#define TOBS 8
#define PREDL 12
#define ATILE (16 * 40)    // one A tile in LDS: 16 rows, pitch 40 bf16 (80B)

typedef __attribute__((ext_vector_type(16))) __bf16 v16bf;
typedef __attribute__((ext_vector_type(8)))  float  v8f;
typedef __attribute__((ext_vector_type(4)))  unsigned int v4u;

union Frag { v16bf bf; v4u u[2]; };

__device__ inline unsigned short f2bf(float x) {        // RNE (cold paths)
    unsigned int u = __float_as_uint(x);
    unsigned int r = u + 0x7FFFu + ((u >> 16) & 1u);
    return (unsigned short)(r >> 16);
}
__device__ inline unsigned short f2bf_t(float x) {      // truncate (hot path)
    return (unsigned short)(__float_as_uint(x) >> 16);  // -> ds_store_b16_d16_hi
}

__device__ inline v8f wmma_bf16(const Frag& A, const Frag& B, v8f C) {
    return __builtin_amdgcn_wmma_f32_16x16x32_bf16(
        false, A.bf, false, B.bf, (short)0, C, false, false);
}

// ---------------------------------------------------------------------------
// Prep: zero state, convert weights to bf16 in WMMA-friendly layouts.
// WC [512][256] bf16 : [W_ih | W_hh] rows (K contiguous -> B fragments)
// WTT [64][2048] bf16 : W_t transposed (K contiguous -> B fragments)
// ---------------------------------------------------------------------------
__global__ __launch_bounds__(256) void k_prep(
    const float* __restrict__ W_ih, const float* __restrict__ W_hh,
    const float* __restrict__ W_t,
    const float* __restrict__ b_ih, const float* __restrict__ b_hh,
    float* __restrict__ C, unsigned short* __restrict__ XH,
    unsigned short* __restrict__ HT,
    unsigned short* __restrict__ WC, unsigned short* __restrict__ WTT,
    float* __restrict__ BI)
{
    int i = blockIdx.x * 256 + threadIdx.x;       // 512*256 = 131072 threads
    if (i < NPED * RNN) {                          // 65536
        C[i]  = 0.f;
        HT[i] = 0;                                 // h0 = 0 (bf16 zero bits)
        XH[(i >> 7) * 256 + 128 + (i & 127)] = 0;  // h part of [ie|te|h]
    }
    if (i < 512 * 256) {                           // WC
        int j = i >> 8, k = i & 255;
        float v = (k < 128) ? W_ih[j * 128 + k] : W_hh[j * 128 + (k - 128)];
        WC[i] = f2bf(v);
    }
    if (i < 64 * 2048) {                           // WTT[e][k] = W_t[k][e]
        int e = i >> 11, k = i & 2047;
        WTT[i] = f2bf(W_t[k * 64 + e]);
    }
    if (i < 512) BI[i] = b_ih[i] + b_hh[i];
}

// ---------------------------------------------------------------------------
// Einsum: 256 WGs x 256 thr (8 waves), TWO peds per WG.
// Wave w owns r-tile [16w,16w+16) and computes C tiles for both peds from a
// single shared B fragment (HT[r][n] bf16, direct global b128 loads).
// A = grid[p]^T staged f32->bf16 (truncation, folds into ds_store_b16_d16_hi)
// into double-buffered LDS (pitch 40 bf16 = 80B, conflict-free b128 reads).
// Each thread stages one float4 (coalesced global_load_b128 of the grid
// stream); one barrier per k-step; stream prefetched 2 chunks ahead.
// Output ST bf16 [512][2048], index g*128+r (matches reshape(N, G2*RNN)).
// ---------------------------------------------------------------------------
__global__ __launch_bounds__(256) void k_einsum(
    const float* __restrict__ grid_t,          // [512][512][16] f32
    const unsigned short* __restrict__ HT,     // [128][512] bf16
    unsigned short* __restrict__ ST)           // [512][2048] bf16
{
    __shared__ __align__(16) unsigned short a_lds[2][2 * ATILE]; // 5 KB

    const int p0   = blockIdx.x * 2;
    const int tid  = threadIdx.x;
    const int lane = tid & 31;
    const int wave = tid >> 5;       // 0..7 -> r tile
    const int half = lane >> 4;
    const int l16  = lane & 15;

    // staging coords: 4 consecutive elements e4..e4+3 of the 1024-elem chunk
    const int e4  = 4 * tid;             // 0..1020
    const int ped = e4 >> 9;             // 0 or 1
    const int el  = e4 & 511;            // within one ped's 32x16 tile
    const int sg  = el & 15;             // multiple of 4 -> rows sg..sg+3
    const int sn  = el >> 4;             // n' within 32-chunk
    const float* gsrc = grid_t + (size_t)(p0 + ped) * (NPED * G2C) + el;
    const int soff = ped * ATILE + sg * 40 + sn;

    v8f acc0 = {0.f, 0.f, 0.f, 0.f, 0.f, 0.f, 0.f, 0.f};
    v8f acc1 = {0.f, 0.f, 0.f, 0.f, 0.f, 0.f, 0.f, 0.f};

    // prologue: stage chunk 0
    {
        float4 v = *(const float4*)gsrc;
        unsigned short* d = a_lds[0] + soff;
        d[0]   = f2bf_t(v.x);  d[40]  = f2bf_t(v.y);
        d[80]  = f2bf_t(v.z);  d[120] = f2bf_t(v.w);
    }
    __syncthreads();

    const unsigned short* hrow =
        HT + (size_t)(wave * 16 + l16) * NPED + half * 8;

    for (int kk = 0; kk < 16; ++kk) {
        const int k0 = kk * 32;

        // shared B fragment: K runs = k0 + half*8 + 0..7 and k0 + 16 + half*8 + 0..7
        Frag B;
        B.u[0] = *(const v4u*)(hrow + k0);
        B.u[1] = *(const v4u*)(hrow + k0 + 16);

        // prefetch grid stream 2 chunks ahead (global_prefetch_b8)
        if (kk + 2 < 16)
            __builtin_prefetch((const void*)(gsrc + (k0 + 64) * G2C), 0, 1);

        // stage next chunk into the other buffer (free since barrier kk-1)
        if (kk + 1 < 16) {
            float4 v = *(const float4*)(gsrc + (k0 + 32) * G2C);
            unsigned short* d = a_lds[(kk + 1) & 1] + soff;
            d[0]   = f2bf_t(v.x);  d[40]  = f2bf_t(v.y);
            d[80]  = f2bf_t(v.z);  d[120] = f2bf_t(v.w);
        }

        // A fragments for both peds: runs K=half*8+0..7 and K=16+half*8+0..7
        const unsigned short* ab = a_lds[kk & 1] + l16 * 40 + half * 8;
        Frag A0, A1;
        A0.u[0] = *(const v4u*)ab;
        A0.u[1] = *(const v4u*)(ab + 16);
        A1.u[0] = *(const v4u*)(ab + ATILE);
        A1.u[1] = *(const v4u*)(ab + ATILE + 16);

        acc0 = wmma_bf16(A0, B, acc0);
        acc1 = wmma_bf16(A1, B, acc1);
        __syncthreads();
    }

    unsigned short* st0 = ST + (size_t)p0 * 2048 + wave * 16 + l16;
    unsigned short* st1 = st0 + 2048;
#pragma unroll
    for (int v = 0; v < 8; ++v) {                 // g = v + 8*half
        st0[(v + 8 * half) * RNN] = f2bf(acc0[v]);
        st1[(v + 8 * half) * RNN] = f2bf(acc1[v]);
    }
}

// ---------------------------------------------------------------------------
// G1: te = relu(ST @ W_t + b_t) via WMMA (M=512,N=64,K=2048) + scalar ie.
// Writes XH[p][0:64]=ie bf16, XH[p][64:128]=te bf16.
// ---------------------------------------------------------------------------
__global__ __launch_bounds__(256) void k_g1(
    const unsigned short* __restrict__ ST,     // [512][2048] bf16
    const unsigned short* __restrict__ WTT,    // [64][2048]  bf16
    const float* __restrict__ b_t,
    const float* __restrict__ pos,             // [512][2]
    const float* __restrict__ W_in,            // [2][64]
    const float* __restrict__ b_in,
    unsigned short* __restrict__ XH)           // [512][256] bf16
{
    const int tid = threadIdx.x;
    // ---- ie (tiny, scalar): exactly 8 elements per thread ----
    const int base = blockIdx.x * 256 + tid;   // 0..4095
#pragma unroll
    for (int it = 0; it < 8; ++it) {
        int i = base + it * 4096;              // covers 512*64
        int p = i >> 6, ec = i & 63;
        float v = pos[p * 2] * W_in[ec] + pos[p * 2 + 1] * W_in[64 + ec]
                + b_in[ec];
        XH[p * 256 + ec] = f2bf(v > 0.f ? v : 0.f);
    }
    // ---- te GEMM ----
    const int lane = tid & 31, wave = tid >> 5;
    const int half = lane >> 4, l16 = lane & 15;
    const int tile = blockIdx.x * 8 + wave;    // 0..127
    const int mt = tile >> 2, nt = tile & 3;

    v8f acc = {0.f, 0.f, 0.f, 0.f, 0.f, 0.f, 0.f, 0.f};
    const unsigned short* ar = ST  + (size_t)(mt * 16 + l16) * 2048;
    const unsigned short* br = WTT + (size_t)(nt * 16 + l16) * 2048;
#pragma unroll 4
    for (int k0 = 0; k0 < 2048; k0 += 32) {
        Frag A, B;
        A.u[0] = *(const v4u*)(ar + k0 + half * 8);
        A.u[1] = *(const v4u*)(ar + k0 + 16 + half * 8);
        B.u[0] = *(const v4u*)(br + k0 + half * 8);
        B.u[1] = *(const v4u*)(br + k0 + 16 + half * 8);
        acc = wmma_bf16(A, B, acc);
    }
    const int ec = nt * 16 + l16;
    const float bt = b_t[ec];
#pragma unroll
    for (int v = 0; v < 8; ++v) {
        int p = mt * 16 + v + 8 * half;
        float t = acc[v] + bt;
        XH[p * 256 + 64 + ec] = f2bf(t > 0.f ? t : 0.f);
    }
}

// ---------------------------------------------------------------------------
// G2: gates = XH @ WC^T + (b_ih+b_hh)   (M=512, N=512, K=256) via WMMA.
// ---------------------------------------------------------------------------
__global__ __launch_bounds__(256) void k_g2(
    const unsigned short* __restrict__ XH,     // [512][256] bf16 = [ie|te|h]
    const unsigned short* __restrict__ WC,     // [512][256] bf16 = [W_ih|W_hh]
    const float* __restrict__ BI,              // [512]
    float* __restrict__ GATES)                 // [512][512]
{
    const int tid = threadIdx.x;
    const int lane = tid & 31, wave = tid >> 5;
    const int half = lane >> 4, l16 = lane & 15;
    const int tile = blockIdx.x * 8 + wave;    // 0..1023
    const int mt = tile >> 5, nt = tile & 31;

    v8f acc = {0.f, 0.f, 0.f, 0.f, 0.f, 0.f, 0.f, 0.f};
    const unsigned short* ar = XH + (size_t)(mt * 16 + l16) * 256;
    const unsigned short* br = WC + (size_t)(nt * 16 + l16) * 256;
#pragma unroll
    for (int k0 = 0; k0 < 256; k0 += 32) {
        Frag A, B;
        A.u[0] = *(const v4u*)(ar + k0 + half * 8);
        A.u[1] = *(const v4u*)(ar + k0 + 16 + half * 8);
        B.u[0] = *(const v4u*)(br + k0 + half * 8);
        B.u[1] = *(const v4u*)(br + k0 + 16 + half * 8);
        acc = wmma_bf16(A, B, acc);
    }
    const int j = nt * 16 + l16;
    const float b = BI[j];
#pragma unroll
    for (int v = 0; v < 8; ++v)
        GATES[(size_t)(mt * 16 + v + 8 * half) * 512 + j] = acc[v] + b;
}

// ---------------------------------------------------------------------------
// LSTM cell (pointwise) + output head. One ped per 128-thread block.
// Writes c, and h in both bf16 layouts consumed next step.
// ---------------------------------------------------------------------------
__global__ __launch_bounds__(128) void k_lstm(
    const float* __restrict__ GATES,
    float* __restrict__ C,
    unsigned short* __restrict__ XH, unsigned short* __restrict__ HT,
    const float* __restrict__ W_out, const float* __restrict__ b_out,
    float* __restrict__ POS, float* __restrict__ out, int pred)
{
    __shared__ float hsh[RNN];
    const int n = blockIdx.x, r = threadIdx.x;
    const float* g = GATES + (size_t)n * 512;
    float ig = g[r], fg = g[128 + r], gg = g[256 + r], og = g[384 + r];
    float c = C[n * RNN + r];
    float si = 1.f / (1.f + expf(-ig));
    float sf = 1.f / (1.f + expf(-fg));
    float so = 1.f / (1.f + expf(-og));
    float c2 = sf * c + si * tanhf(gg);
    float h2 = so * tanhf(c2);
    C[n * RNN + r] = c2;
    XH[n * 256 + 128 + r] = f2bf(h2);   // A-side layout for gate GEMM
    HT[r * NPED + n]      = f2bf(h2);   // B-side layout for einsum

    if (pred) {
        hsh[r] = h2;
        __syncthreads();
        if (r < OUTC) {
            float a = b_out[r];
            for (int k = 0; k < RNN; ++k) a += hsh[k] * W_out[k * OUTC + r];
            out[n * OUTC + r] = a;
            if (r < 2) POS[n * 2 + r] = a;   // next-step position input
        }
    }
}

// ---------------------------------------------------------------------------
extern "C" void kernel_launch(void* const* d_in, const int* in_sizes, int n_in,
                              void* d_out, int out_size, void* d_ws, size_t ws_size,
                              hipStream_t stream) {
    const float* obs   = (const float*)d_in[0];   // [1,8,512,2]
    const float* grids = (const float*)d_in[1];   // [1,8,512,512,16]
    const float* W_in  = (const float*)d_in[2];
    const float* b_in  = (const float*)d_in[3];
    const float* W_t   = (const float*)d_in[4];
    const float* b_t   = (const float*)d_in[5];
    const float* W_ih  = (const float*)d_in[6];
    const float* W_hh  = (const float*)d_in[7];
    const float* b_ih  = (const float*)d_in[8];
    const float* b_hh  = (const float*)d_in[9];
    const float* W_out = (const float*)d_in[10];
    const float* b_out = (const float*)d_in[11];
    float* out = (float*)d_out;

    // workspace layout (~4.3 MB)
    char* ws = (char*)d_ws;
    float*          C_    = (float*)(ws);                              // 256KB
    unsigned short* HT    = (unsigned short*)(ws + (256 << 10));       // 128KB
    unsigned short* XH    = (unsigned short*)(ws + (384 << 10));       // 256KB
    unsigned short* ST    = (unsigned short*)(ws + (640 << 10));       // 2MB
    float*          GATES = (float*)(ws + (2688 << 10));               // 1MB
    unsigned short* WC    = (unsigned short*)(ws + (3712 << 10));      // 256KB
    unsigned short* WTT   = (unsigned short*)(ws + (3968 << 10));      // 256KB
    float*          BI    = (float*)(ws + (4224 << 10));               // 2KB
    float*          POS   = (float*)(ws + (4226 << 10));               // 4KB

    k_prep<<<512, 256, 0, stream>>>(W_ih, W_hh, W_t, b_ih, b_hh,
                                    C_, XH, HT, WC, WTT, BI);

    for (int s = 0; s < TOBS + PREDL; ++s) {
        const int t = (s < TOBS) ? s : (TOBS - 1);
        const float* grid_t = grids + (size_t)t * NPED * NPED * G2C;
        const float* pos = (s < TOBS)  ? (obs + (size_t)s * NPED * 2)
                         : (s == TOBS) ? (obs + (size_t)(TOBS - 1) * NPED * 2)
                                       : POS;
        const int pred = (s >= TOBS);
        float* outp = out + (size_t)(pred ? (s - TOBS) : 0) * NPED * OUTC;

        k_einsum<<<NPED / 2, 256, 0, stream>>>(grid_t, HT, ST);
        k_g1<<<16, 256, 0, stream>>>(ST, WTT, b_t, pos, W_in, b_in, XH);
        k_g2<<<128, 256, 0, stream>>>(XH, WC, BI, GATES);
        k_lstm<<<NPED, 128, 0, stream>>>(GATES, C_, XH, HT,
                                         W_out, b_out, POS, outp, pred);
    }
}
